// PipelinedMoEBlock_46617575031027
// MI455X (gfx1250) — matrix operand order repaired and checked by reading, not compiled, not run on hardware
//
#include <hip/hip_runtime.h>
#include <hip/hip_bf16.h>
#include <stdint.h>

// ---------------------------------------------------------------------------
// PipelinedMoEBlock for MI455X (gfx1250): bf16 WMMA + async/TDM LDS staging.
//   B=2 S=512 D=768 H=12 HD=64 E=8 K(top)=2 F=3072
// ---------------------------------------------------------------------------

#define DEV __device__ __forceinline__

constexpr int Bq = 2, Sq = 512, Dq = 768, Hq = 12, HDq = 64, Eq = 8, Fq = 3072;
constexpr int NT = Bq * Sq;   // 1024 tokens

typedef __bf16 bf16_t;
typedef bf16_t v16bf __attribute__((ext_vector_type(16)));
typedef float  v8f   __attribute__((ext_vector_type(8)));
typedef int    v4i_t __attribute__((ext_vector_type(4)));   // b128 async payload

union FragBF { v16bf v; unsigned int u[8]; };

// CDNA5 async / TDM availability (toolchain-dependent; fall back to sync path)
#if __has_builtin(__builtin_amdgcn_global_load_async_to_lds_b128) && \
    __has_builtin(__builtin_amdgcn_s_wait_asynccnt)
#define HAVE_ASYNC 1
#else
#define HAVE_ASYNC 0
#endif
#if __has_builtin(__builtin_amdgcn_tensor_load_to_lds) && \
    __has_builtin(__builtin_amdgcn_s_wait_tensorcnt) && HAVE_ASYNC
#define HAVE_TDM 1
#else
#define HAVE_TDM 0
#endif

#define AS1 __attribute__((address_space(1)))
#define AS3 __attribute__((address_space(3)))

// fp32 -> bf16 with round-to-nearest-even
DEV unsigned short f2bf(float f) {
  unsigned int x = __float_as_uint(f);
  return (unsigned short)((x + 0x7fffu + ((x >> 16) & 1u)) >> 16);
}

DEV float gelu_tanh(float v) {
  return 0.5f * v * (1.0f + tanhf(0.7978845608028654f * (v + 0.044715f * v * v * v)));
}

// A-fragment dword index within a 16-dword (32-half) row, per ISA 7.12.2
// "16-bit A-Matrix 16x32": lanes 0-15 hold K 0..7 (VGPR0-3), 16..23 (VGPR4-7);
// lanes 16-31 are +8 in K.
DEV int a_frag_dword(int j, int group) {
  return ((j < 4) ? j : (j + 4)) + 4 * group;
}

// ---------------------------------------------------------------------------
__global__ void k_f32_to_bf16(const float* __restrict__ in,
                              unsigned short* __restrict__ out, int n) {
  int i = blockIdx.x * blockDim.x + threadIdx.x;
  if (i < n) out[i] = f2bf(in[i]);
}

// ---------------------------------------------------------------------------
// LayerNorm over D=768; one block (256 thr) per token. Writes bf16 (+opt f32).
// ---------------------------------------------------------------------------
__global__ void k_layernorm(const float* __restrict__ x, const float* __restrict__ w,
                            const float* __restrict__ b,
                            unsigned short* __restrict__ obf, float* __restrict__ of) {
  __shared__ float s1[256], s2[256];
  const int tok = blockIdx.x, t = threadIdx.x;
  const float* row = x + (size_t)tok * Dq;
  float a = 0.f, q = 0.f;
  for (int i = t; i < Dq; i += 256) { float v = row[i]; a += v; q += v * v; }
  s1[t] = a; s2[t] = q; __syncthreads();
  for (int s = 128; s > 0; s >>= 1) {
    if (t < s) { s1[t] += s1[t + s]; s2[t] += s2[t + s]; }
    __syncthreads();
  }
  const float mu   = s1[0] * (1.0f / Dq);
  const float var  = s2[0] * (1.0f / Dq) - mu * mu;
  const float rstd = rsqrtf(var + 1e-5f);
  for (int i = t; i < Dq; i += 256) {
    float v = (row[i] - mu) * rstd * w[i] + b[i];
    obf[(size_t)tok * Dq + i] = f2bf(v);
    if (of) of[(size_t)tok * Dq + i] = v;
  }
}

// ---------------------------------------------------------------------------
// Tiled bf16 WMMA GEMM:  C[M,N] = act(A[M,K] @ B[K,N] + bias) (+ epilogues)
// Block 256 thr = 8 waves; tile 128x64; wave = 32x32 (2x2 accumulators).
// K staged in 32-chunks, double-buffered LDS via async-LDS loads (A) and a
// TDM 2D tile descriptor with padding (B) when the toolchain exposes them.
//   MODE 0: Cf = v + bias
//   MODE 1: Cf = v + bias + resid
//   MODE 2: Ch = bf16(gelu(v + bias))
//   MODE 3: Cf += rowscale[row*rsStride] * (v + bias)     (accumulate)
// Requires: M % 128 == 0, N % 64 == 0, K % 32 == 0 (true for all uses here).
// ---------------------------------------------------------------------------
constexpr int A_DW = 20;   // A row stride in dwords (40 halfs, 80B, 16B-aligned)
constexpr int B_DW = 36;   // B row stride in dwords (72 halfs, 144B, 16B-aligned)

template <int MODE>
__global__ void k_gemm_bf16(const unsigned short* __restrict__ A,
                            const unsigned short* __restrict__ Bm,
                            const float* __restrict__ bias,
                            float* __restrict__ Cf, unsigned short* __restrict__ Ch,
                            const float* __restrict__ resid,
                            const float* __restrict__ rowscale, int rsStride,
                            int M, int N, int K) {
  __shared__ __align__(16) unsigned short As[2][128 * (2 * A_DW)];
  __shared__ __align__(16) unsigned short Bs[2][32 * (2 * B_DW)];

  const int t = threadIdx.x;
  const int lane = t & 31, wave = t >> 5;
  const int wn = wave & 1, wm = wave >> 1;          // 2 N-waves x 4 M-waves
  const int group = lane >> 4, r15 = lane & 15;
  const int mBase = blockIdx.y * 128;
  const int nBase = blockIdx.x * 64;
  const int nk = K / 32;

  // staging geometry: A tile 128 rows x 16 dw (each thread: 2x b128);
  //                   B tile  32 rows x 32 dw (each thread: 1x b128)
  const int aRow = t >> 1, aCd = (t & 1) * 8;
  const int bRow = t >> 3, bCd = (t & 7) * 4;
  const unsigned short* aSrc = A + (size_t)(mBase + aRow) * K + aCd * 2;
  const unsigned short* bSrc = Bm + (size_t)bRow * N + nBase + bCd * 2;

  v8f acc[4]{};

  auto computeChunk = [&](const unsigned short* Ab, const unsigned short* Bb) {
    const unsigned int* Asd = (const unsigned int*)Ab;
    const unsigned int* Bsd = (const unsigned int*)Bb;
    FragBF b0, b1, a0, a1;
#pragma unroll
    for (int j = 0; j < 8; ++j) {
      const int dj = a_frag_dword(j, group);
      b0.u[j] = (unsigned)Bb[(2 * dj) * (2 * B_DW) + wn * 32 + r15] |
                ((unsigned)Bb[(2 * dj + 1) * (2 * B_DW) + wn * 32 + r15] << 16);
      b1.u[j] = (unsigned)Bb[(2 * dj) * (2 * B_DW) + wn * 32 + 16 + r15] |
                ((unsigned)Bb[(2 * dj + 1) * (2 * B_DW) + wn * 32 + 16 + r15] << 16);
      a0.u[j] = Asd[(wm * 32 +  0 + r15) * A_DW + dj];
      a1.u[j] = Asd[(wm * 32 + 16 + r15) * A_DW + dj];
    }
    acc[0] = __builtin_amdgcn_wmma_f32_16x16x32_bf16(false, a0.v, false, b0.v,
                                                     (short)0, acc[0], false, false);
    acc[1] = __builtin_amdgcn_wmma_f32_16x16x32_bf16(false, a0.v, false, b1.v,
                                                     (short)0, acc[1], false, false);
    acc[2] = __builtin_amdgcn_wmma_f32_16x16x32_bf16(false, a1.v, false, b0.v,
                                                     (short)0, acc[2], false, false);
    acc[3] = __builtin_amdgcn_wmma_f32_16x16x32_bf16(false, a1.v, false, b1.v,
                                                     (short)0, acc[3], false, false);
  };

#if HAVE_ASYNC
  unsigned short* aDst[2] = { &As[0][aRow * 2 * A_DW + aCd * 2],
                              &As[1][aRow * 2 * A_DW + aCd * 2] };
#if !HAVE_TDM
  unsigned short* bDst[2] = { &Bs[0][bRow * 2 * B_DW + bCd * 2],
                              &Bs[1][bRow * 2 * B_DW + bCd * 2] };
#endif
#if HAVE_TDM
  typedef unsigned int v4u __attribute__((ext_vector_type(4)));
  typedef int v4i __attribute__((ext_vector_type(4)));
  typedef int v8i __attribute__((ext_vector_type(8)));
  // D# group1: data_size=2B, pad_enable, pad_interval=32dw, pad_amount=4dw,
  // tensor_dim0=64, tensor_dim1=32, tile_dim0=64, tile_dim1=32, stride0=N.
  const v8i tdmG1 = { (int)((1u << 16) | (1u << 20) | (4u << 22) | (3u << 25)),
                      (int)(64u << 16), (int)(32u << 16), (int)(64u << 16),
                      32, (int)N, 0, 0 };
  const v4i z4 = {0, 0, 0, 0};
#if __clang_major__ >= 23
  const v8i z8 = {0, 0, 0, 0, 0, 0, 0, 0};
#endif
  const unsigned bLds[2] = {
      (unsigned)(size_t)(AS3 unsigned short*)&Bs[0][0],
      (unsigned)(size_t)(AS3 unsigned short*)&Bs[1][0] };
#define ASYNC_INFLIGHT 2
#else
#define ASYNC_INFLIGHT 3
#endif

  auto stage = [&](int ic, int buf) {
    const unsigned short* ag = aSrc + ic * 32;
    __builtin_amdgcn_global_load_async_to_lds_b128(
        (AS1 v4i_t*)ag, (AS3 v4i_t*)aDst[buf], 0, 0);
    __builtin_amdgcn_global_load_async_to_lds_b128(
        (AS1 v4i_t*)ag, (AS3 v4i_t*)aDst[buf], 16, 0);
#if HAVE_TDM
    if (wave == 0) {
      unsigned long long ga = (unsigned long long)(size_t)(bSrc - bCd * 2 -
                                  (size_t)bRow * N + (size_t)ic * 32 * N);
      v4u g0 = { 1u, bLds[buf], (unsigned)ga,
                 ((unsigned)((ga >> 32) & 0x01FFFFFFu)) | (2u << 30) };
#if __clang_major__ >= 23
      __builtin_amdgcn_tensor_load_to_lds(g0, tdmG1, z4, z4, z8, 0);
#else
      __builtin_amdgcn_tensor_load_to_lds(g0, tdmG1, z4, z4, 0);
#endif
    }
#else
    __builtin_amdgcn_global_load_async_to_lds_b128(
        (AS1 v4i_t*)(bSrc + (size_t)ic * 32 * N), (AS3 v4i_t*)bDst[buf], 0, 0);
#endif
  };

  stage(0, 0);
  for (int ic = 0; ic < nk; ++ic) {
    const int buf = ic & 1;
    if (ic + 1 < nk) {
      stage(ic + 1, buf ^ 1);
      __builtin_amdgcn_s_wait_asynccnt(ASYNC_INFLIGHT);
#if HAVE_TDM
      if (wave == 0) __builtin_amdgcn_s_wait_tensorcnt(1);
#endif
    } else {
      __builtin_amdgcn_s_wait_asynccnt(0);
#if HAVE_TDM
      if (wave == 0) __builtin_amdgcn_s_wait_tensorcnt(0);
#endif
    }
    __syncthreads();
    computeChunk(As[buf], Bs[buf]);
    __syncthreads();
  }
#else   // ---- sync fallback staging ----
  for (int ic = 0; ic < nk; ++ic) {
    const int kb = ic * 32;
    uint4 av0 = *(const uint4*)(aSrc + kb);
    uint4 av1 = *(const uint4*)(aSrc + kb + 8);
    uint4 bv  = *(const uint4*)(bSrc + (size_t)kb * N);
    unsigned int* Ad = (unsigned int*)As[0] + aRow * A_DW + aCd;
    Ad[0] = av0.x; Ad[1] = av0.y; Ad[2] = av0.z; Ad[3] = av0.w;
    Ad[4] = av1.x; Ad[5] = av1.y; Ad[6] = av1.z; Ad[7] = av1.w;
    unsigned int* Bd = (unsigned int*)Bs[0] + bRow * B_DW + bCd;
    Bd[0] = bv.x; Bd[1] = bv.y; Bd[2] = bv.z; Bd[3] = bv.w;
    __syncthreads();
    computeChunk(As[0], Bs[0]);
    __syncthreads();
  }
#endif

  // epilogue: 2x2 accumulators, C layout M = r + 8*group, N = lane&15
#pragma unroll
  for (int ns = 0; ns < 2; ++ns) {
    const int gCol = nBase + wn * 32 + ns * 16 + r15;
    const float bb = bias[gCol];
#pragma unroll
    for (int ms = 0; ms < 2; ++ms) {
      const v8f a = acc[ms * 2 + ns];
#pragma unroll
      for (int r = 0; r < 8; ++r) {
        const int gRow = mBase + wm * 32 + ms * 16 + r + 8 * group;
        const size_t ci = (size_t)gRow * N + gCol;
        float v = a[r] + bb;
        if (MODE == 0)      Cf[ci] = v;
        else if (MODE == 1) Cf[ci] = v + resid[ci];
        else if (MODE == 2) Ch[ci] = f2bf(gelu_tanh(v));
        else                Cf[ci] += rowscale[(size_t)gRow * rsStride] * v;
      }
    }
  }
}

// ---------------------------------------------------------------------------
// Split qkv[1024,2304] (fp32) into per-head bf16: q[BH,S,HD], kT[BH,HD,S], v[BH,S,HD]
// ---------------------------------------------------------------------------
__global__ void k_split_qkv(const float* __restrict__ qkv,
                            unsigned short* __restrict__ q,
                            unsigned short* __restrict__ kT,
                            unsigned short* __restrict__ v) {
  int i = blockIdx.x * blockDim.x + threadIdx.x;
  if (i >= Bq * Hq * Sq * HDq) return;
  const int d = i & 63, s = (i >> 6) & 511, bh = i >> 15;
  const int b = bh / Hq, h = bh % Hq;
  const float* base = qkv + (size_t)(b * Sq + s) * (3 * Dq) + h * HDq + d;
  q [((size_t)bh * Sq + s) * HDq + d] = f2bf(base[0]);
  kT[((size_t)bh * HDq + d) * Sq + s] = f2bf(base[Dq]);
  v [((size_t)bh * Sq + s) * HDq + d] = f2bf(base[2 * Dq]);
}

// ---------------------------------------------------------------------------
// Attention: one 32-thread block per (b,h,16-row q tile).
// scores (16x512) via WMMA -> LDS, two-pass softmax, P@V via WMMA.
// ---------------------------------------------------------------------------
__global__ void k_attention(const unsigned short* __restrict__ qb,
                            const unsigned short* __restrict__ kTb,
                            const unsigned short* __restrict__ vb,
                            unsigned short* __restrict__ obf) {
  __shared__ __align__(16) float sc[16][Sq + 4];
  __shared__ __align__(16) unsigned short pb[16][Sq];
  const int lane = threadIdx.x;
  const int group = lane >> 4, r15 = lane & 15;
  const int qt = blockIdx.x, h = blockIdx.y, b = blockIdx.z;
  const int bh = b * Hq + h;

  // Q A-fragments for the two d-chunks (d 0..31, 32..63)
  const unsigned int* qd = (const unsigned int*)(qb + ((size_t)bh * Sq + qt * 16) * HDq);
  FragBF qa[2];
#pragma unroll
  for (int c = 0; c < 2; ++c)
#pragma unroll
    for (int j = 0; j < 8; ++j)
      qa[c].u[j] = qd[r15 * 32 + c * 16 + a_frag_dword(j, group)];

  // scores
  const unsigned int* kTd = (const unsigned int*)kTb;
  for (int kt = 0; kt < Sq / 16; ++kt) {
    v8f s{};
#pragma unroll
    for (int c = 0; c < 2; ++c) {
      FragBF kf;   // B-fragment: lane%16 = key column, elements = d in A-fragment k order
#pragma unroll
      for (int j = 0; j < 8; ++j)
        kf.u[j] = (unsigned)kTb[((size_t)bh * HDq + c * 32 + 2 * a_frag_dword(j, group)) * Sq + kt * 16 + r15] |
                  ((unsigned)kTb[((size_t)bh * HDq + c * 32 + 2 * a_frag_dword(j, group) + 1) * Sq + kt * 16 + r15] << 16);
      s = __builtin_amdgcn_wmma_f32_16x16x32_bf16(false, qa[c].v, false, kf.v,
                                                  (short)0, s, false, false);
    }
#pragma unroll
    for (int r = 0; r < 8; ++r)
      sc[r + 8 * group][kt * 16 + r15] = s[r] * 0.125f;   // 1/sqrt(64)
  }
  __syncthreads();

  // softmax: lane handles row=r15, half=group (256 cols each); combine via shfl
  {
    const int row = r15, c0 = group * 256;
    float m = -3.0e38f;
    for (int c = 0; c < 256; ++c) m = fmaxf(m, sc[row][c0 + c]);
    m = fmaxf(m, __shfl_xor(m, 16, 32));
    float sum = 0.f;
    for (int c = 0; c < 256; ++c) {
      float e = __expf(sc[row][c0 + c] - m);
      sc[row][c0 + c] = e; sum += e;
    }
    sum += __shfl_xor(sum, 16, 32);
    const float inv = 1.0f / sum;
    for (int c = 0; c < 256; ++c) pb[row][c0 + c] = f2bf(sc[row][c0 + c] * inv);
  }
  __syncthreads();

  // O = P @ V  (contraction over keys in chunks of 32; 4 HD tiles of 16)
  v8f o[4]{};
  const unsigned int* pbu = (const unsigned int*)pb;
  const unsigned int* vd  = (const unsigned int*)vb;
  for (int kc = 0; kc < Sq / 32; ++kc) {
    FragBF pf;
#pragma unroll
    for (int j = 0; j < 8; ++j)
      pf.u[j] = pbu[r15 * (Sq / 2) + kc * 16 + a_frag_dword(j, group)];
#pragma unroll
    for (int nt = 0; nt < 4; ++nt) {
      FragBF vf;   // B-fragment: lane%16 = d column, elements = keys in A-fragment k order
#pragma unroll
      for (int j = 0; j < 8; ++j)
        vf.u[j] = (unsigned)vb[((size_t)bh * Sq + kc * 32 + 2 * a_frag_dword(j, group)) * HDq + nt * 16 + r15] |
                  ((unsigned)vb[((size_t)bh * Sq + kc * 32 + 2 * a_frag_dword(j, group) + 1) * HDq + nt * 16 + r15] << 16);
      o[nt] = __builtin_amdgcn_wmma_f32_16x16x32_bf16(false, pf.v, false, vf.v,
                                                      (short)0, o[nt], false, false);
    }
  }
#pragma unroll
  for (int nt = 0; nt < 4; ++nt)
#pragma unroll
    for (int r = 0; r < 8; ++r) {
      const int qrow = qt * 16 + r + 8 * group;
      obf[((size_t)(b * Sq + qrow)) * Dq + h * HDq + nt * 16 + r15] = f2bf(o[nt][r]);
    }
}

// ---------------------------------------------------------------------------
// Gate: logits = moe_in @ gate_w; top-2 softmax -> dense comb[NT, E]
// ---------------------------------------------------------------------------
__global__ void k_gate(const float* __restrict__ xin, const float* __restrict__ gw,
                       float* __restrict__ comb) {
  const int tkn = blockIdx.x * blockDim.x + threadIdx.x;
  if (tkn >= NT) return;
  float lg[Eq];
#pragma unroll
  for (int e = 0; e < Eq; ++e) lg[e] = 0.f;
  const float* row = xin + (size_t)tkn * Dq;
  for (int d = 0; d < Dq; ++d) {
    const float xv = row[d];
#pragma unroll
    for (int e = 0; e < Eq; ++e) lg[e] += xv * gw[d * Eq + e];
  }
  int i1 = 0; float v1 = lg[0];
  for (int e = 1; e < Eq; ++e) if (lg[e] > v1) { v1 = lg[e]; i1 = e; }
  int i2 = 0; float v2 = -3.0e38f;
  for (int e = 0; e < Eq; ++e) if (e != i1 && lg[e] > v2) { v2 = lg[e]; i2 = e; }
  const float e2 = __expf(v2 - v1);
  const float g1 = 1.0f / (1.0f + e2), g2 = e2 / (1.0f + e2);
#pragma unroll
  for (int e = 0; e < Eq; ++e) comb[(size_t)tkn * Eq + e] = 0.f;
  comb[(size_t)tkn * Eq + i1] = g1;
  comb[(size_t)tkn * Eq + i2] = g2;
}

// ---------------------------------------------------------------------------
extern "C" void kernel_launch(void* const* d_in, const int* in_sizes, int n_in,
                              void* d_out, int out_size, void* d_ws, size_t ws_size,
                              hipStream_t stream) {
  (void)in_sizes; (void)n_in; (void)out_size; (void)ws_size;
  const float* x     = (const float*)d_in[0];
  const float* ln1w  = (const float*)d_in[1];
  const float* ln1b  = (const float*)d_in[2];
  const float* ln2w  = (const float*)d_in[3];
  const float* ln2b  = (const float*)d_in[4];
  const float* wqkv  = (const float*)d_in[5];
  const float* bqkv  = (const float*)d_in[6];
  const float* wout  = (const float*)d_in[7];
  const float* bout  = (const float*)d_in[8];
  const float* gatew = (const float*)d_in[9];
  const float* w1    = (const float*)d_in[10];
  const float* b1    = (const float*)d_in[11];
  const float* w2    = (const float*)d_in[12];
  const float* b2    = (const float*)d_in[13];
  float* out = (float*)d_out;

  // workspace carve-out (~46 MB), 256B aligned slices
  char* wsp = (char*)d_ws;
  auto alloc = [&](size_t bytes) {
    char* p = wsp; wsp += (bytes + 255) & ~(size_t)255; return p;
  };
  unsigned short* xn_bf   = (unsigned short*)alloc((size_t)NT * Dq * 2);
  unsigned short* wqkv_bf = (unsigned short*)alloc((size_t)Dq * 3 * Dq * 2);
  float*          qkv_f   = (float*)alloc((size_t)NT * 3 * Dq * 4);
  unsigned short* q_bf    = (unsigned short*)alloc((size_t)NT * Dq * 2);
  unsigned short* kT_bf   = (unsigned short*)alloc((size_t)NT * Dq * 2);
  unsigned short* v_bf    = (unsigned short*)alloc((size_t)NT * Dq * 2);
  unsigned short* o_bf    = (unsigned short*)alloc((size_t)NT * Dq * 2);
  unsigned short* wout_bf = (unsigned short*)alloc((size_t)Dq * Dq * 2);
  float*          xres_f  = (float*)alloc((size_t)NT * Dq * 4);
  float*          xn2_f   = (float*)alloc((size_t)NT * Dq * 4);
  unsigned short* xn2_bf  = (unsigned short*)alloc((size_t)NT * Dq * 2);
  float*          comb    = (float*)alloc((size_t)NT * Eq * 4);
  unsigned short* w1e_bf  = (unsigned short*)alloc((size_t)Dq * Fq * 2);
  unsigned short* w2e_bf  = (unsigned short*)alloc((size_t)Fq * Dq * 2);
  unsigned short* h_bf    = (unsigned short*)alloc((size_t)NT * Fq * 2);

  // 1) LN1 -> bf16
  k_layernorm<<<NT, 256, 0, stream>>>(x, ln1w, ln1b, xn_bf, nullptr);
  // 2) QKV projection
  { int n = Dq * 3 * Dq;
    k_f32_to_bf16<<<(n + 255) / 256, 256, 0, stream>>>(wqkv, wqkv_bf, n); }
  k_gemm_bf16<0><<<dim3(3 * Dq / 64, NT / 128), 256, 0, stream>>>(
      xn_bf, wqkv_bf, bqkv, qkv_f, nullptr, nullptr, nullptr, 0, NT, 3 * Dq, Dq);
  // 3) split to per-head bf16 (K transposed for contiguous B-fragments)
  { int n = Bq * Hq * Sq * HDq;
    k_split_qkv<<<(n + 255) / 256, 256, 0, stream>>>(qkv_f, q_bf, kT_bf, v_bf); }
  // 4) attention
  k_attention<<<dim3(Sq / 16, Hq, Bq), 32, 0, stream>>>(q_bf, kT_bf, v_bf, o_bf);
  // 5) out projection + residual
  { int n = Dq * Dq;
    k_f32_to_bf16<<<(n + 255) / 256, 256, 0, stream>>>(wout, wout_bf, n); }
  k_gemm_bf16<1><<<dim3(Dq / 64, NT / 128), 256, 0, stream>>>(
      o_bf, wout_bf, bout, xres_f, nullptr, x, nullptr, 0, NT, Dq, Dq);
  // 6) LN2 (bf16 for GEMMs, fp32 for gating)
  k_layernorm<<<NT, 256, 0, stream>>>(xres_f, ln2w, ln2b, xn2_bf, xn2_f);
  // 7) top-2 gate
  k_gate<<<(NT + 255) / 256, 256, 0, stream>>>(xn2_f, gatew, comb);
  // 8) out := x_resid, then accumulate gated expert outputs
  (void)hipMemcpyAsync(out, xres_f, (size_t)NT * Dq * 4,
                       hipMemcpyDeviceToDevice, stream);
  for (int e = 0; e < Eq; ++e) {
    { int n = Dq * Fq;
      k_f32_to_bf16<<<(n + 255) / 256, 256, 0, stream>>>(w1 + (size_t)e * n, w1e_bf, n); }
    k_gemm_bf16<2><<<dim3(Fq / 64, NT / 128), 256, 0, stream>>>(
        xn2_bf, w1e_bf, b1 + (size_t)e * Fq, nullptr, h_bf, nullptr, nullptr, 0,
        NT, Fq, Dq);
    { int n = Fq * Dq;
      k_f32_to_bf16<<<(n + 255) / 256, 256, 0, stream>>>(w2 + (size_t)e * n, w2e_bf, n); }
    k_gemm_bf16<3><<<dim3(Dq / 64, NT / 128), 256, 0, stream>>>(
        h_bf, w2e_bf, b2 + (size_t)e * Dq, out, nullptr, nullptr, comb + e, Eq,
        NT, Dq, Fq);
  }
}
